// RouteNet_Fermi_wavelet_single_level_34462817583672
// MI455X (gfx1250) — matrix operand order repaired and checked by reading, not compiled, not run on hardware
//
#include <hip/hip_runtime.h>

// ---------------------------------------------------------------------------
// RouteNet-Fermi forward on MI455X (gfx1250, wave32, WMMA bf16 16x16x32).
//
// Sizes (fixed by reference): F=100000 P=8 T=32 L=10000 K=80 H=32 ITERS=8
//
// GEMM stages run as v_wmma_f32_16x16x32_bf16 with fp32 accumulators:
//   - embedding GRU over T=32 (combined [x|h] @ W, K padded to 64)
//   - path GRU over P=8, 8 message-passing iterations (K=96)
//   - readout MLP layers 1&2
// Small L=10000 GRU cells / MLPs stay on VALU.
// __launch_bounds__(128,1) keeps the 8 f32 accumulator tiles register-
// resident (default occupancy budget was spilling them to scratch).
// ---------------------------------------------------------------------------

#define F_N   100000
#define P_N   8
#define T_N   32
#define L_N   10000
#define KG_N  80
#define H_N   32
#define ITERS 8
#define NW    4            // waves per block in WMMA kernels

typedef __attribute__((ext_vector_type(16))) __bf16 v16bf;
typedef __attribute__((ext_vector_type(8)))  __bf16 v8bf;
typedef __attribute__((ext_vector_type(8)))  float  v8f;

__device__ __forceinline__ float sigmoidf_(float x) {
  return 1.0f / (1.0f + __expf(-x));
}

// exp-based tanh: cheap v_exp_f32 TRANS path, clamped so exp cannot overflow.
__device__ __forceinline__ float tanhf_(float x) {
  float xc = fminf(fmaxf(x, -15.0f), 15.0f);
  float e  = __expf(2.0f * xc);
  return (e - 1.0f) / (e + 1.0f);
}

__device__ __forceinline__ v8f wmma_bf16(v16bf a, v16bf b, v8f c) {
  // (neg_a, A, neg_b, B, c_mod, C, reuse_a, reuse_b)
  return __builtin_amdgcn_wmma_f32_16x16x32_bf16(false, a, false, b, (short)0, c,
                                                 false, false);
}

// Build a 16x32 bf16 A fragment (16-bit A layout) from a staging matrix of
// fp32 rows [16][kstride] held in LDS.  chunk c covers K = 32c..32c+31.
// Lane layout: M = lane&15; lanes>=16 shift K by +8; elements 0..7 and 8..15
// are two contiguous 8-element runs.
__device__ __forceinline__ v16bf build_A(const float* st, int lane, int c,
                                         int kstride) {
  int m    = lane & 15;
  int koff = (lane >> 4) * 8;
  const float* row = st + m * kstride + c * 32 + koff;
  v16bf a;
#pragma unroll
  for (int e = 0; e < 8; ++e) a[e] = (__bf16)row[e];
#pragma unroll
  for (int e = 0; e < 8; ++e) a[8 + e] = (__bf16)row[16 + e];
  return a;
}

// Load a 32x16 bf16 B fragment for n-tile j, k-chunk c from packed transposed
// weights WT[n][k] (row-major, kstride columns) in LDS.
// Lane layout: N = lane&15; lanes>=16 shift K by +8; elements 0..7 and 8..15
// are K runs at +0 and +16 (matches build_A).
__device__ __forceinline__ v16bf load_B(const __bf16* w, int lane, int j, int c,
                                        int kstride) {
  int n  = j * 16 + (lane & 15);
  int kb = c * 32 + ((lane < 16) ? 0 : 8);
  const __bf16* p = w + n * kstride + kb;
  v8bf lo = *(const v8bf*)(p);
  v8bf hi = *(const v8bf*)(p + 16);
  return __builtin_shufflevector(lo, hi, 0, 1, 2, 3, 4, 5, 6, 7,
                                 8, 9, 10, 11, 12, 13, 14, 15);
}

// GRU gate math entirely in the C/D fragment layout.
// acc tiles: [0,1]=z  [2,3]=r  [4,5]=xh(+b0h)  [6,7]=hh(+b1h)
__device__ __forceinline__ void gru_gates(const v8f* acc, v8f* hold) {
#pragma unroll
  for (int j = 0; j < 2; ++j) {
    v8f hn;
#pragma unroll
    for (int e = 0; e < 8; ++e) {
      float z  = sigmoidf_(acc[j][e]);
      float r  = sigmoidf_(acc[2 + j][e]);
      float hc = tanhf_(acc[4 + j][e] + r * acc[6 + j][e]);
      hn[e] = z * hold[j][e] + (1.0f - z) * hc;
    }
    hold[j] = hn;
  }
}

// Scatter h (D layout) back into staging columns [hbase, hbase+32).
__device__ __forceinline__ void store_h_D(float* st, int lane, const v8f* hold,
                                          int hbase, int kstride) {
  int m16 = lane & 15;
  int lh  = (lane >> 4) * 8;
#pragma unroll
  for (int j = 0; j < 2; ++j)
#pragma unroll
    for (int e = 0; e < 8; ++e)
      st[(e + lh) * kstride + hbase + j * 16 + m16] = hold[j][e];
}

// ---------------------------------------------------------------------------
// Weight packing: combined GRU weight WT[n][k] (bf16, transposed) + bias.
//   columns n: [0,64)=z,r (K+R rows, bias b0+b1)  [64,96)=xh (K rows, b0h)
//              [96,128)=hh (R rows, b1h)
//   rows k: [0,din)=K rows, [din,din+32)=R rows, rest zero padding.
// ---------------------------------------------------------------------------
__global__ void pack_gru_kernel(const float* __restrict__ Kp,
                                const float* __restrict__ Rp,
                                const float* __restrict__ b, __bf16* WT,
                                float* biasC, int din, int kpad) {
  int i = blockIdx.x * blockDim.x + threadIdx.x;
  if (i >= 128 * kpad) return;
  int n = i / kpad, k = i % kpad;
  float v = 0.0f;
  if (n < 64) {
    if (k < din) v = Kp[k * 96 + n];
    else if (k < din + 32) v = Rp[(k - din) * 96 + n];
  } else if (n < 96) {
    if (k < din) v = Kp[k * 96 + n];
  } else {
    if (k >= din && k < din + 32) v = Rp[(k - din) * 96 + (n - 32)];
  }
  WT[n * kpad + k] = (__bf16)v;
  if (k == 0) {
    float bb;
    if (n < 64)      bb = b[n] + b[96 + n];
    else if (n < 96) bb = b[n];
    else             bb = b[96 + (n - 32)];
    biasC[n] = bb;
  }
}

// Transpose-pack a (din x 16) dense weight into WT[16][32] bf16 (zero-padded K).
__global__ void pack_ro_kernel(const float* __restrict__ W, __bf16* WT, int din) {
  int i = blockIdx.x * blockDim.x + threadIdx.x;
  if (i >= 16 * 32) return;
  int n = i / 32, k = i % 32;
  WT[n * 32 + k] = (k < din) ? (__bf16)W[k * 16 + n] : (__bf16)0.0f;
}

// ---------------------------------------------------------------------------
// Embedding GRU: T=32 steps, input [ps,ipg] (2) + h (32), K padded to 64.
// One wave per 16-flow tile.  Writes final h -> path_state.
// ---------------------------------------------------------------------------
__global__ void __launch_bounds__(NW * 32, 1)
emb_gru_kernel(const float* __restrict__ ps, const float* __restrict__ ipg,
               const __bf16* __restrict__ WTg, const float* __restrict__ biasg,
               float* path_state, int ntiles) {
  __shared__ __align__(32) __bf16 w[128 * 64];
  __shared__ __align__(32) float  stg[NW][16 * 64];
  for (int i = threadIdx.x; i < 128 * 64; i += blockDim.x) w[i] = WTg[i];
  __syncthreads();
  int wave = threadIdx.x >> 5, lane = threadIdx.x & 31;
  int tile = blockIdx.x * NW + wave;
  if (tile >= ntiles) return;
  int    fbase = tile * 16;
  float* st    = stg[wave];
  for (int i = lane; i < 16 * 64; i += 32) st[i] = 0.0f;  // pad + h(0)=0
  int m = lane & 15;
  v8f hold[2];
#pragma unroll
  for (int j = 0; j < 2; ++j)
#pragma unroll
    for (int e = 0; e < 8; ++e) hold[j][e] = 0.0f;
  float bias[8];
#pragma unroll
  for (int j = 0; j < 8; ++j) bias[j] = biasg[j * 16 + m];

  for (int t = 0; t < T_N; ++t) {
    st[m * 64 + 0] = ps[(fbase + m) * T_N + t];
    st[m * 64 + 1] = ipg[(fbase + m) * T_N + t];
    __builtin_amdgcn_wave_barrier();
    v16bf A0 = build_A(st, lane, 0, 64);
    v16bf A1 = build_A(st, lane, 1, 64);
    v8f acc[8];
#pragma unroll
    for (int j = 0; j < 8; ++j) {
#pragma unroll
      for (int e = 0; e < 8; ++e) acc[j][e] = bias[j];
    }
#pragma unroll
    for (int j = 0; j < 8; ++j) {
      acc[j] = wmma_bf16(A0, load_B(w, lane, j, 0, 64), acc[j]);
      acc[j] = wmma_bf16(A1, load_B(w, lane, j, 1, 64), acc[j]);
    }
    gru_gates(acc, hold);
    store_h_D(st, lane, hold, 2, 64);
    __builtin_amdgcn_wave_barrier();
  }
  int fl = lane >> 1, col = (lane & 1) * 16;
#pragma unroll
  for (int e = 0; e < 16; ++e)
    path_state[(fbase + fl) * 32 + col + e] = st[fl * 64 + 2 + col + e];
}

// ---------------------------------------------------------------------------
// Path GRU: 8 steps, input [queue_gather|link_gather] (64) + h (32), K=96.
// Writes seq[:,0,:]=prev h, seq[:,p+1,:]=h_p, and final h -> path_state.
// ---------------------------------------------------------------------------
__global__ void __launch_bounds__(NW * 32, 1)
path_gru_kernel(const float* __restrict__ qstate,
                const float* __restrict__ lstate, const int* __restrict__ l2p,
                float* path_state, float* seq, const __bf16* __restrict__ WTg,
                const float* __restrict__ biasg, int ntiles) {
  __shared__ __align__(32) __bf16 w[128 * 96];
  __shared__ __align__(32) float  stg[NW][16 * 96];
  for (int i = threadIdx.x; i < 128 * 96; i += blockDim.x) w[i] = WTg[i];
  __syncthreads();
  int wave = threadIdx.x >> 5, lane = threadIdx.x & 31;
  int tile = blockIdx.x * NW + wave;
  if (tile >= ntiles) return;
  int    fbase = tile * 16;
  float* st    = stg[wave];
  int m   = lane & 15;
  int lh  = (lane >> 4) * 8;    // element row offset (D layout)
  int lc  = (lane >> 4) * 16;   // column half (gathers)

  // initial h -> registers (D layout) + seq[:,0,:] coalesced copy
  v8f hold[2];
#pragma unroll
  for (int j = 0; j < 2; ++j)
#pragma unroll
    for (int e = 0; e < 8; ++e)
      hold[j][e] = path_state[(fbase + e + lh) * 32 + j * 16 + m];
  store_h_D(st, lane, hold, 64, 96);
  {
    int fl = lane >> 1, col = (lane & 1) * 16;
#pragma unroll
    for (int e = 0; e < 16; ++e) {
      float v = path_state[(fbase + fl) * 32 + col + e];
      seq[((long)(fbase + fl) * 9 + 0) * 32 + col + e] = v;
    }
  }
  float bias[8];
#pragma unroll
  for (int j = 0; j < 8; ++j) bias[j] = biasg[j * 16 + m];

  for (int p = 0; p < P_N; ++p) {
    // gather queue/link state rows for this hop into staging cols [0,64)
    int li = l2p[(fbase + m) * P_N + p];
#pragma unroll
    for (int e = 0; e < 16; ++e)
      st[m * 96 + lc + e] = qstate[li * 32 + lc + e];
#pragma unroll
    for (int e = 0; e < 16; ++e)
      st[m * 96 + 32 + lc + e] = lstate[li * 32 + lc + e];
    __builtin_amdgcn_wave_barrier();

    v16bf A0 = build_A(st, lane, 0, 96);
    v16bf A1 = build_A(st, lane, 1, 96);
    v16bf A2 = build_A(st, lane, 2, 96);
    v8f acc[8];
#pragma unroll
    for (int j = 0; j < 8; ++j) {
#pragma unroll
      for (int e = 0; e < 8; ++e) acc[j][e] = bias[j];
    }
#pragma unroll
    for (int j = 0; j < 8; ++j) {
      acc[j] = wmma_bf16(A0, load_B(w, lane, j, 0, 96), acc[j]);
      acc[j] = wmma_bf16(A1, load_B(w, lane, j, 1, 96), acc[j]);
      acc[j] = wmma_bf16(A2, load_B(w, lane, j, 2, 96), acc[j]);
    }
    gru_gates(acc, hold);
    store_h_D(st, lane, hold, 64, 96);
    __builtin_amdgcn_wave_barrier();
    int fl = lane >> 1, col = (lane & 1) * 16;
#pragma unroll
    for (int e = 0; e < 16; ++e)
      seq[((long)(fbase + fl) * 9 + (p + 1)) * 32 + col + e] =
          st[fl * 96 + 64 + col + e];
  }
  int fl = lane >> 1, col = (lane & 1) * 16;
#pragma unroll
  for (int e = 0; e < 16; ++e)
    path_state[(fbase + fl) * 32 + col + e] = st[fl * 96 + 64 + col + e];
}

// ---------------------------------------------------------------------------
// Readout MLP over (F*8, 32) rows: relu(x@W1+b1) -> relu(@W2+b2) -> @W3+b3.
// ---------------------------------------------------------------------------
__global__ void __launch_bounds__(NW * 32, 1)
readout_kernel(const float* __restrict__ seq, const __bf16* __restrict__ W1Tg,
               const __bf16* __restrict__ W2Tg, const float* __restrict__ b1g,
               const float* __restrict__ b2g, const float* __restrict__ W3g,
               const float* __restrict__ b3g, float* occ, int ntiles) {
  __shared__ __align__(32) __bf16 w1[16 * 32];
  __shared__ __align__(32) __bf16 w2[16 * 32];
  __shared__ float w3[16];
  __shared__ __align__(32) float stg[NW][16 * 32];
  for (int i = threadIdx.x; i < 16 * 32; i += blockDim.x) {
    w1[i] = W1Tg[i];
    w2[i] = W2Tg[i];
  }
  if (threadIdx.x < 16) w3[threadIdx.x] = W3g[threadIdx.x];
  __syncthreads();
  int wave = threadIdx.x >> 5, lane = threadIdx.x & 31;
  int tile = blockIdx.x * NW + wave;
  if (tile >= ntiles) return;
  float* st  = stg[wave];
  int m    = lane & 15;
  int lh   = (lane >> 4) * 8;
  int koff = lh;
  int rid  = tile * 16 + m;
  int f = rid >> 3, p = rid & 7;
  const float* src = seq + ((long)f * 9 + (p + 1)) * 32;
  v16bf A;
#pragma unroll
  for (int e = 0; e < 8; ++e) A[e] = (__bf16)src[koff + e];
#pragma unroll
  for (int e = 0; e < 8; ++e) A[8 + e] = (__bf16)src[16 + koff + e];
  v8f acc1;
#pragma unroll
  for (int e = 0; e < 8; ++e) acc1[e] = b1g[m];
  acc1 = wmma_bf16(A, load_B(w1, lane, 0, 0, 32), acc1);
#pragma unroll
  for (int e = 0; e < 8; ++e) {
    st[(e + lh) * 32 + m]      = fmaxf(acc1[e], 0.0f);
    st[(e + lh) * 32 + 16 + m] = 0.0f;  // K padding must be exact zero
  }
  __builtin_amdgcn_wave_barrier();
  v16bf A2 = build_A(st, lane, 0, 32);
  v8f acc2;
#pragma unroll
  for (int e = 0; e < 8; ++e) acc2[e] = b2g[m];
  acc2 = wmma_bf16(A2, load_B(w2, lane, 0, 0, 32), acc2);
#pragma unroll
  for (int e = 0; e < 8; ++e) st[(e + lh) * 32 + m] = fmaxf(acc2[e], 0.0f);
  __builtin_amdgcn_wave_barrier();
  if (lane < 16) {
    float s = b3g[0];
#pragma unroll
    for (int c = 0; c < 16; ++c) s += st[lane * 32 + c] * w3[c];
    occ[tile * 16 + lane] = s;
  }
}

// ---------------------------------------------------------------------------
// Small VALU kernels for the L=10000-sized pieces.
// ---------------------------------------------------------------------------
__global__ void load_kernel(const float* __restrict__ ft,
                            const float* __restrict__ cap,
                            const int* __restrict__ p2l, float* loadv) {
  int l = blockIdx.x * blockDim.x + threadIdx.x;
  if (l >= L_N) return;
  float s = 0.0f;
  for (int k = 0; k < KG_N; ++k) s += ft[p2l[(l * KG_N + k) * 2]];
  loadv[l] = s / cap[l];
}

__global__ void onehot_kernel(const int* __restrict__ bt, float* oh) {
  int l = blockIdx.x * blockDim.x + threadIdx.x;
  if (l >= L_N) return;
  oh[l * 2 + 0] = (bt[l] == 0) ? 1.0f : 0.0f;
  oh[l * 2 + 1] = (bt[l] == 1) ? 1.0f : 0.0f;
}

__global__ void dense_relu_kernel(const float* __restrict__ X,
                                  const float* __restrict__ W,
                                  const float* __restrict__ b, float* Y,
                                  int rows, int din, int dout) {
  int i = blockIdx.x * blockDim.x + threadIdx.x;
  if (i >= rows * dout) return;
  int r = i / dout, c = i % dout;
  float s = b[c];
  for (int k = 0; k < din; ++k) s += X[r * din + k] * W[k * dout + c];
  Y[i] = fmaxf(s, 0.0f);
}

__global__ void gru_cell_kernel(const float* __restrict__ x,
                                const float* __restrict__ h,
                                const float* __restrict__ Kp,
                                const float* __restrict__ Rp,
                                const float* __restrict__ b, float* out,
                                int rows) {
  int i = blockIdx.x * blockDim.x + threadIdx.x;
  if (i >= rows * 32) return;
  int r = i >> 5, c = i & 31;
  float xz = b[c], xr = b[32 + c], xh = b[64 + c];
  float hz = b[96 + c], hr = b[128 + c], hh = b[160 + c];
  for (int k = 0; k < 32; ++k) {
    float xv = x[r * 32 + k], hv = h[r * 32 + k];
    xz += xv * Kp[k * 96 + c];
    xr += xv * Kp[k * 96 + 32 + c];
    xh += xv * Kp[k * 96 + 64 + c];
    hz += hv * Rp[k * 96 + c];
    hr += hv * Rp[k * 96 + 32 + c];
    hh += hv * Rp[k * 96 + 64 + c];
  }
  float z = sigmoidf_(xz + hz), rr = sigmoidf_(xr + hr);
  float hc = tanhf_(xh + rr * hh);
  out[i] = z * h[i] + (1.0f - z) * hc;
}

__global__ void psum_kernel(const float* __restrict__ seq,
                            const int* __restrict__ p2l, float* psum) {
  int i = blockIdx.x * blockDim.x + threadIdx.x;
  if (i >= L_N * 32) return;
  int l = i >> 5, c = i & 31;
  float s = 0.0f;
  for (int k = 0; k < KG_N; ++k) {
    int fi  = p2l[(l * KG_N + k) * 2];
    int pos = p2l[(l * KG_N + k) * 2 + 1];
    s += seq[((long)fi * 9 + pos) * 32 + c];
  }
  psum[i] = s;
}

__global__ void final_kernel(const float* __restrict__ occ,
                             const float* __restrict__ cap,
                             const int* __restrict__ l2p,
                             const float* __restrict__ ft,
                             const float* __restrict__ fp, float* out) {
  int f = blockIdx.x * blockDim.x + threadIdx.x;
  if (f >= F_N) return;
  float qd = 0.0f, invs = 0.0f;
  for (int p = 0; p < P_N; ++p) {
    float c = cap[l2p[f * P_N + p]] * 1.0e9f;
    qd   += occ[f * P_N + p] / c;
    invs += 1.0f / c;
  }
  out[f] = qd + (ft[f] / fp[f]) * invs;
}

// ---------------------------------------------------------------------------
extern "C" void kernel_launch(void* const* d_in, const int* in_sizes, int n_in,
                              void* d_out, int out_size, void* d_ws,
                              size_t ws_size, hipStream_t stream) {
  (void)in_sizes; (void)n_in; (void)out_size; (void)ws_size;
  // Inputs: setup_inputs() order, params flattened by sorted key.
  const float* ft     = (const float*)d_in[0];
  const float* fpk    = (const float*)d_in[1];
  /* d_in[2] flow_length unused */
  const float* cap    = (const float*)d_in[3];
  const int*   bt     = (const int*)d_in[4];
  const int*   l2p    = (const int*)d_in[5];
  const int*   p2l    = (const int*)d_in[6];
  /* d_in[7] queue_to_link == arange(L), unused */
  const float* pswt   = (const float*)d_in[8];
  const float* ipgwt  = (const float*)d_in[9];
  const float* emb_K  = (const float*)d_in[10];
  const float* emb_R  = (const float*)d_in[11];
  const float* emb_b  = (const float*)d_in[12];
  const float* l_K    = (const float*)d_in[13];
  const float* l_R    = (const float*)d_in[14];
  const float* l_b    = (const float*)d_in[15];
  const float* le_W1  = (const float*)d_in[16];
  const float* le_W2  = (const float*)d_in[17];
  const float* le_b1  = (const float*)d_in[18];
  const float* le_b2  = (const float*)d_in[19];
  const float* path_K = (const float*)d_in[20];
  const float* path_R = (const float*)d_in[21];
  const float* path_b = (const float*)d_in[22];
  const float* q_K    = (const float*)d_in[23];
  const float* q_R    = (const float*)d_in[24];
  const float* q_b    = (const float*)d_in[25];
  const float* qe_W1  = (const float*)d_in[26];
  const float* qe_W2  = (const float*)d_in[27];
  const float* qe_b1  = (const float*)d_in[28];
  const float* qe_b2  = (const float*)d_in[29];
  const float* ro_W1  = (const float*)d_in[30];
  const float* ro_W2  = (const float*)d_in[31];
  const float* ro_W3  = (const float*)d_in[32];
  const float* ro_b1  = (const float*)d_in[33];
  const float* ro_b2  = (const float*)d_in[34];
  const float* ro_b3  = (const float*)d_in[35];
  float* out = (float*)d_out;

  // Workspace carve-up (256B aligned).
  char*  ws  = (char*)d_ws;
  size_t off = 0;
  auto carve = [&](size_t bytes) {
    size_t o = off;
    off = (off + bytes + 255) & ~(size_t)255;
    return o;
  };
  __bf16* wt_emb   = (__bf16*)(ws + carve(128 * 64 * 2));
  float*  b_emb    = (float*)(ws + carve(128 * 4));
  __bf16* wt_path  = (__bf16*)(ws + carve(128 * 96 * 2));
  float*  b_path   = (float*)(ws + carve(128 * 4));
  __bf16* w1t      = (__bf16*)(ws + carve(16 * 32 * 2));
  __bf16* w2t      = (__bf16*)(ws + carve(16 * 32 * 2));
  float*  pstate   = (float*)(ws + carve((size_t)F_N * 32 * 4));
  float*  seq      = (float*)(ws + carve((size_t)F_N * 9 * 32 * 4));
  float*  loadv    = (float*)(ws + carve((size_t)L_N * 4));
  float*  oneh     = (float*)(ws + carve((size_t)L_N * 2 * 4));
  float*  h1tmp    = (float*)(ws + carve((size_t)L_N * 32 * 4));
  float*  qA       = (float*)(ws + carve((size_t)L_N * 32 * 4));
  float*  qB       = (float*)(ws + carve((size_t)L_N * 32 * 4));
  float*  lA       = (float*)(ws + carve((size_t)L_N * 32 * 4));
  float*  lB       = (float*)(ws + carve((size_t)L_N * 32 * 4));
  float*  psum     = (float*)(ws + carve((size_t)L_N * 32 * 4));
  float*  occ      = (float*)(ws + carve((size_t)F_N * P_N * 4));

  const int NT_F  = F_N / 16;            // 6250 flow tiles (exact)
  const int NT_RO = (F_N * P_N) / 16;    // 50000 readout tiles (exact)
  const int BLK   = NW * 32;             // 128 threads, 4 waves

  // 1) pack weights to bf16 transposed layouts
  pack_gru_kernel<<<(128 * 64 + 255) / 256, 256, 0, stream>>>(
      emb_K, emb_R, emb_b, wt_emb, b_emb, 2, 64);
  pack_gru_kernel<<<(128 * 96 + 255) / 256, 256, 0, stream>>>(
      path_K, path_R, path_b, wt_path, b_path, 64, 96);
  pack_ro_kernel<<<2, 256, 0, stream>>>(ro_W1, w1t, 32);
  pack_ro_kernel<<<2, 256, 0, stream>>>(ro_W2, w2t, 16);

  // 2) embedding GRU -> initial path_state
  emb_gru_kernel<<<(NT_F + NW - 1) / NW, BLK, 0, stream>>>(pswt, ipgwt, wt_emb,
                                                           b_emb, pstate, NT_F);

  // 3) link & queue state init
  load_kernel<<<(L_N + 255) / 256, 256, 0, stream>>>(ft, cap, p2l, loadv);
  onehot_kernel<<<(L_N + 255) / 256, 256, 0, stream>>>(bt, oneh);
  dense_relu_kernel<<<(L_N * 32 + 255) / 256, 256, 0, stream>>>(
      loadv, le_W1, le_b1, h1tmp, L_N, 1, 32);
  dense_relu_kernel<<<(L_N * 32 + 255) / 256, 256, 0, stream>>>(
      h1tmp, le_W2, le_b2, lA, L_N, 32, 32);
  dense_relu_kernel<<<(L_N * 32 + 255) / 256, 256, 0, stream>>>(
      oneh, qe_W1, qe_b1, h1tmp, L_N, 2, 32);
  dense_relu_kernel<<<(L_N * 32 + 255) / 256, 256, 0, stream>>>(
      h1tmp, qe_W2, qe_b2, qA, L_N, 32, 32);

  // 4) message-passing iterations
  float *qcur = qA, *qnxt = qB, *lcur = lA, *lnxt = lB;
  for (int it = 0; it < ITERS; ++it) {
    path_gru_kernel<<<(NT_F + NW - 1) / NW, BLK, 0, stream>>>(
        qcur, lcur, l2p, pstate, seq, wt_path, b_path, NT_F);
    psum_kernel<<<(L_N * 32 + 255) / 256, 256, 0, stream>>>(seq, p2l, psum);
    gru_cell_kernel<<<(L_N * 32 + 255) / 256, 256, 0, stream>>>(
        psum, qcur, q_K, q_R, q_b, qnxt, L_N);
    gru_cell_kernel<<<(L_N * 32 + 255) / 256, 256, 0, stream>>>(
        qnxt, lcur, l_K, l_R, l_b, lnxt, L_N);
    float* t;
    t = qcur; qcur = qnxt; qnxt = t;
    t = lcur; lcur = lnxt; lnxt = t;
  }

  // 5) readout MLP + final delay combine
  readout_kernel<<<NT_RO / NW, BLK, 0, stream>>>(seq, w1t, w2t, ro_b1, ro_b2,
                                                 ro_W3, ro_b3, occ, NT_RO);
  final_kernel<<<(F_N + 255) / 256, 256, 0, stream>>>(occ, cap, l2p, ft, fpk,
                                                      out);
}
